// MultiWindowAttention_85615878078712
// MI455X (gfx1250) — matrix operand order repaired and checked by reading, not compiled, not run on hardware
//
#include <hip/hip_runtime.h>
#include <hip/hip_bf16.h>

typedef __attribute__((ext_vector_type(16))) _Float16 v16h;
typedef __attribute__((ext_vector_type(8)))  _Float16 v8h;
typedef __attribute__((ext_vector_type(2)))  _Float16 v2h;
typedef __attribute__((ext_vector_type(8)))  float    v8f;
typedef __attribute__((ext_vector_type(4)))  float    v4f;

#define B_   4
#define L_   2048
#define H_   8
#define E_   64
#define D_   64
#define SCALE 0.125f        // 1/sqrt(64), folded into Q at load time
#define NEGINF (-1e30f)
#define VT_STRIDE 40        // 32 keys + 8 halfs pad (80B rows -> 16B aligned, bank spread)

#define WMMA16(A, Bm, C) \
  __builtin_amdgcn_wmma_f32_16x16x32_f16(false, (A), false, (Bm), (short)0, (C), false, false)

// 16 consecutive f32 -> v16h  (K B-fragment element order: e -> K = hs*16 + e)
__device__ __forceinline__ v16h load_cvt16(const float* p) {
  const v4f* q = (const v4f*)p;
  v4f a = q[0], b = q[1], c = q[2], d = q[3];
  v16h r;
#pragma unroll
  for (int i = 0; i < 4; ++i) {
    r[i]      = (_Float16)a[i];
    r[4 + i]  = (_Float16)b[i];
    r[8 + i]  = (_Float16)c[i];
    r[12 + i] = (_Float16)d[i];
  }
  return r;
}

// two separate 8-f32 chunks, pre-scaled -> v16h (A-frag: elems 0..7 = chunk0, 8..15 = chunk1)
__device__ __forceinline__ v16h load_cvt8x2_scaled(const float* p0, const float* p1, float s) {
  const v4f* q0 = (const v4f*)p0;
  const v4f* q1 = (const v4f*)p1;
  v4f a = q0[0], b = q0[1], c = q1[0], d = q1[1];
  v16h r;
#pragma unroll
  for (int i = 0; i < 4; ++i) {
    r[i]      = (_Float16)(a[i] * s);
    r[4 + i]  = (_Float16)(b[i] * s);
    r[8 + i]  = (_Float16)(c[i] * s);
    r[12 + i] = (_Float16)(d[i] * s);
  }
  return r;
}

// band mask + online-max + exp + P stage for one 16-row tile (scores pre-scaled via Q)
#define SM_TILE(S0, S1, QTB, RMX, ALF, TSEL)                                   \
  {                                                                            \
    _Pragma("unroll")                                                          \
    for (int r = 0; r < 8; ++r) {                                              \
      const int i  = (QTB) + r + hs * 8;                                       \
      const int c0 = i - (kb + ln) + w;   /* band <=> (unsigned)c <= 2w */     \
      const int c1 = c0 - 16;                                                  \
      float u0 = ((unsigned)c0 <= w2) ? S0[r] : NEGINF;                        \
      float u1 = ((unsigned)c1 <= w2) ? S1[r] : NEGINF;                        \
      float t = fmaxf(u0, u1);                                                 \
      t = fmaxf(t, __shfl_xor(t, 1, 32));                                      \
      t = fmaxf(t, __shfl_xor(t, 2, 32));                                      \
      t = fmaxf(t, __shfl_xor(t, 4, 32));                                      \
      t = fmaxf(t, __shfl_xor(t, 8, 32));                                      \
      const float mnew = fmaxf(RMX[r], t);                                     \
      ALF[r] = __expf(RMX[r] - mnew); /* ==1 if block fully masked */          \
      RMX[r] = mnew;                                                           \
      const float e0 = __expf(u0 - mnew);                                      \
      const float e1 = __expf(u1 - mnew);                                      \
      const int m = r + hs * 8;                                                \
      Pb[wid][TSEL][m * 32 + ln]      = (_Float16)e0;                          \
      Pb[wid][TSEL][m * 32 + 16 + ln] = (_Float16)e1;                          \
    }                                                                          \
  }

// one 16-dim output tile: load V B-fragment once, feed both row tiles
#define PV2(NT, OA, OB)                                                        \
  {                                                                            \
    const int d_ = (NT) * 16 + ln;                                             \
    v8h va_ = *(const v8h*)&VT[d_ * VT_STRIDE + hs * 8];                       \
    v8h vc_ = *(const v8h*)&VT[d_ * VT_STRIDE + 16 + hs * 8];                  \
    v16h vb_;                                                                  \
    for (int i_ = 0; i_ < 8; ++i_) { vb_[i_] = va_[i_]; vb_[8+i_] = vc_[i_]; } \
    OA = WMMA16(paA, vb_, OA);                                                 \
    OB = WMMA16(paB, vb_, OB);                                                 \
  }

__global__ __launch_bounds__(64, 1) void mwa_flash_wmma(
    const float* __restrict__ Qg, const float* __restrict__ Kg,
    const float* __restrict__ Vg, float* __restrict__ Og) {

  __shared__ __align__(16) _Float16 VT[D_ * VT_STRIDE];   // V^T tile: [dim][key]
  __shared__ __align__(16) _Float16 Pb[2][2][16 * 32];    // [wave][rowtile] P, row-major

  const int tid  = threadIdx.x;
  const int wid  = tid >> 5;    // 0..1
  const int lane = tid & 31;
  const int hs   = lane >> 4;   // half-wave select
  const int ln   = lane & 15;

  const int wg = blockIdx.x;
  const int qt = wg & 31;            // L/64 = 32 query tiles of 64 rows
  const int h  = (wg >> 5) & 7;
  const int b  = wg >> 8;
  const int w  = 32 << (h & 3);      // per-head window
  const unsigned w2 = (unsigned)(2 * w);

  const int qwg = qt << 6;           // first query row of workgroup
  const int qA  = qwg + (wid << 5);  // wave's first row tile
  const int qB  = qA + 16;           // wave's second row tile

  // ---- Q A-fragments (pre-scaled by 1/sqrt(E)) for both row tiles ----
  const float* qrA = Qg + (((long)b * L_ + (qA + ln)) * H_ + h) * E_;
  const float* qrB = Qg + (((long)b * L_ + (qB + ln)) * H_ + h) * E_;
  const v16h qaA0 = load_cvt8x2_scaled(qrA + hs * 16,      qrA + 8 + hs * 16, SCALE);
  const v16h qaA1 = load_cvt8x2_scaled(qrA + 32 + hs * 16, qrA + 40 + hs * 16, SCALE);
  const v16h qaB0 = load_cvt8x2_scaled(qrB + hs * 16,      qrB + 8 + hs * 16, SCALE);
  const v16h qaB1 = load_cvt8x2_scaled(qrB + 32 + hs * 16, qrB + 40 + hs * 16, SCALE);

  // ones B-fragment: rsum rides the P*V WMMA as an extra all-ones V column.
  v16h ones;
#pragma unroll
  for (int i = 0; i < 16; ++i) ones[i] = (_Float16)1.0f;

  v8f oa0 = {}, oa1 = {}, oa2 = {}, oa3 = {}, oa4 = {};
  v8f ob0 = {}, ob1 = {}, ob2 = {}, ob3 = {}, ob4 = {};
  float rmaxA[8], rmaxB[8];
#pragma unroll
  for (int r = 0; r < 8; ++r) { rmaxA[r] = NEGINF; rmaxB[r] = NEGINF; }

  int kstart = qwg - w;        if (kstart < 0) kstart = 0;
  int kend   = qwg + 64 + w;   if (kend > L_) kend = L_;
  // both are multiples of 32

  const int vkp = tid >> 2;    // 0..15: key pair
  const int vgp = tid & 3;     // 0..3 : group of 16 dims

  for (int kb = kstart; kb < kend; kb += 32) {
    __syncthreads();  // protect VT reuse from previous iteration
    // ---- cooperatively stage V^T (32 keys x 64 dims) into LDS as f16 ----
    {
      const int k0 = kb + vkp * 2;
      const v4f* r0 = (const v4f*)(Vg + (((long)b * L_ + k0)     * H_ + h) * D_ + vgp * 16);
      const v4f* r1 = (const v4f*)(Vg + (((long)b * L_ + k0 + 1) * H_ + h) * D_ + vgp * 16);
      v4f A0 = r0[0], A1 = r0[1], A2 = r0[2], A3 = r0[3];
      v4f B0 = r1[0], B1 = r1[1], B2 = r1[2], B3 = r1[3];
#pragma unroll
      for (int i = 0; i < 16; ++i) {
        float x = (i < 4) ? A0[i] : (i < 8) ? A1[i - 4] : (i < 12) ? A2[i - 8] : A3[i - 12];
        float y = (i < 4) ? B0[i] : (i < 8) ? B1[i - 4] : (i < 12) ? B2[i - 8] : B3[i - 12];
        v2h pk; pk[0] = (_Float16)x; pk[1] = (_Float16)y;
        *(v2h*)&VT[(vgp * 16 + i) * VT_STRIDE + vkp * 2] = pk;
      }
    }
    __syncthreads();

    // ---- S = (scaled Q) K^T: K fragments loaded once, used by BOTH row tiles ----
    v8f sA0 = {}, sA1 = {}, sB0 = {}, sB1 = {};
    {
      const float* kr0 = Kg + (((long)b * L_ + (kb + ln))      * H_ + h) * E_;
      const float* kr1 = Kg + (((long)b * L_ + (kb + 16 + ln)) * H_ + h) * E_;
      v16h b0lo = load_cvt16(kr0 + hs * 16);
      v16h b0hi = load_cvt16(kr0 + 32 + hs * 16);
      v16h b1lo = load_cvt16(kr1 + hs * 16);
      v16h b1hi = load_cvt16(kr1 + 32 + hs * 16);
      sA0 = WMMA16(qaA0, b0lo, sA0);  sA0 = WMMA16(qaA1, b0hi, sA0);
      sA1 = WMMA16(qaA0, b1lo, sA1);  sA1 = WMMA16(qaA1, b1hi, sA1);
      sB0 = WMMA16(qaB0, b0lo, sB0);  sB0 = WMMA16(qaB1, b0hi, sB0);
      sB1 = WMMA16(qaB0, b1lo, sB1);  sB1 = WMMA16(qaB1, b1hi, sB1);
    }

    // ---- online softmax for both row tiles ----
    float alfA[8], alfB[8];
    SM_TILE(sA0, sA1, qA, rmaxA, alfA, 0)
    SM_TILE(sB0, sB1, qB, rmaxB, alfB, 1)
#pragma unroll
    for (int r = 0; r < 8; ++r) {
      oa0[r] *= alfA[r]; oa1[r] *= alfA[r]; oa2[r] *= alfA[r];
      oa3[r] *= alfA[r]; oa4[r] *= alfA[r];
      ob0[r] *= alfB[r]; ob1[r] *= alfB[r]; ob2[r] *= alfB[r];
      ob3[r] *= alfB[r]; ob4[r] *= alfB[r];
    }

    // per-wave LDS store -> load ordering (DS ops are in-order per wave)
    asm volatile("s_wait_dscnt 0" ::: "memory");

    // ---- P A-fragments (16x32 f16) from LDS for both row tiles ----
    v16h paA, paB;
    {
      v8h lo = *(const v8h*)&Pb[wid][0][ln * 32 + hs * 8];
      v8h hi = *(const v8h*)&Pb[wid][0][ln * 32 + 16 + hs * 8];
#pragma unroll
      for (int i = 0; i < 8; ++i) { paA[i] = lo[i]; paA[8 + i] = hi[i]; }
      v8h lo2 = *(const v8h*)&Pb[wid][1][ln * 32 + hs * 8];
      v8h hi2 = *(const v8h*)&Pb[wid][1][ln * 32 + 16 + hs * 8];
#pragma unroll
      for (int i = 0; i < 8; ++i) { paB[i] = lo2[i]; paB[8 + i] = hi2[i]; }
    }

    // ---- O += P*V (8 WMMAs) and rsum += P*1 (2 WMMAs); V frags shared ----
    PV2(0, oa0, ob0)
    PV2(1, oa1, ob1)
    PV2(2, oa2, ob2)
    PV2(3, oa3, ob3)
    oa4 = WMMA16(paA, ones, oa4);
    ob4 = WMMA16(paB, ones, ob4);
  }

  // ---- normalize rows and store fp32 output [B,L,H,D] ----
#pragma unroll
  for (int r = 0; r < 8; ++r) {
    const int iA = qA + r + hs * 8;
    const float invA = 1.0f / oa4[r];
    float* orA = Og + (((long)b * L_ + iA) * H_ + h) * D_;
    orA[ln]      = oa0[r] * invA;
    orA[16 + ln] = oa1[r] * invA;
    orA[32 + ln] = oa2[r] * invA;
    orA[48 + ln] = oa3[r] * invA;
    const int iB = qB + r + hs * 8;
    const float invB = 1.0f / ob4[r];
    float* orB = Og + (((long)b * L_ + iB) * H_ + h) * D_;
    orB[ln]      = ob0[r] * invB;
    orB[16 + ln] = ob1[r] * invB;
    orB[32 + ln] = ob2[r] * invB;
    orB[48 + ln] = ob3[r] * invB;
  }
}

extern "C" void kernel_launch(void* const* d_in, const int* in_sizes, int n_in,
                              void* d_out, int out_size, void* d_ws, size_t ws_size,
                              hipStream_t stream) {
  const float* Q = (const float*)d_in[0];
  const float* K = (const float*)d_in[1];
  const float* V = (const float*)d_in[2];
  float* O = (float*)d_out;
  (void)in_sizes; (void)n_in; (void)out_size; (void)d_ws; (void)ws_size;
  dim3 grid(B_ * H_ * (L_ / 64));   // 1024 workgroups, 2 waves x 32 rows each
  mwa_flash_wmma<<<grid, dim3(64), 0, stream>>>(Q, K, V, O);
}
